// fused_classSR_3class_rcan_net_70377334112292
// MI455X (gfx1250) — matrix operand order repaired and checked by reading, not compiled, not run on hardware
//
#include <hip/hip_runtime.h>

// ---------------------------------------------------------------------------
// ClassSR 3-class RCAN on gfx1250 (MI455X): implicit-GEMM conv via
// v_wmma_f32_16x16x32_f16.  f16 activations/weights, f32 accumulate.
// Block = 4 waves (128 thr): 64(Mchan) x 64(Npix) tile.
//  - double-buffered LDS im2col B tile, ONE barrier per K-stage
//  - fragment-major LDS layout (k contiguous, 80B row stride) ->
//    ds_load_b128 fragment reads / ds_store_b128 staging
//  - 16 WMMAs per block per K-stage (4 per wave, A fragment reused 4x)
//  - im2col k->(c,r,s) decomposition precomputed in a per-layer table,
//    fetched via wave-uniform SCALAR loads (readfirstlane forces SMEM path);
//    border test = 1 shift+and vs a per-thread 9-bit validity mask.
//  - weights pre-packed [O][Kpad] f16 zero-padded -> unguarded b128 A loads
//  - pow2 spatial dims -> shift/mask pixel decomposition (no divisions)
// ---------------------------------------------------------------------------

typedef __attribute__((ext_vector_type(16))) _Float16 v16h;
typedef __attribute__((ext_vector_type(8)))  _Float16 v8h;
typedef __attribute__((ext_vector_type(8)))  float    v8f;

template<int KH, int KW, int PAD>
__global__ void __launch_bounds__(128)
conv_wmma_kernel(const _Float16* __restrict__ X,    // [N,C,H,W] f16
                 const _Float16* __restrict__ Wt,   // [O][Kpad] f16 zero-padded
                 const int2*     __restrict__ Tab,  // [Kpad] {doff, rs} table
                 const float*    __restrict__ Bs,   // [O] bias f32
                 const _Float16* __restrict__ Res,  // optional residual f16
                 _Float16*       __restrict__ Yh,   // f16 out (nullable)
                 float*          __restrict__ Yf,   // f32 out (nullable)
                 int N, int C, int logH, int logW, int O, int relu)
{
    constexpr int KHW = KH * KW;
    const int tid   = threadIdx.x;
    const int lane  = tid & 31;
    const int wv    = tid >> 5;        // wave id 0..3 -> M sub-tile
    const int hsel  = lane >> 4;       // lane half
    const int l16   = lane & 15;
    const int logHW = logH + logW;
    const int H     = 1 << logH;
    const int W     = 1 << logW;
    const int HW    = 1 << logHW;
    const int P     = N << logHW;      // total pixels
    const int K     = C * KHW;
    const int Kp    = (K + 31) & ~31;  // padded K (weight row stride)
    const int m0    = (blockIdx.x << 6) + (wv << 4);
    const int p0    = blockIdx.y << 6;

    // fragment-major B tile: [pixel][k], k contiguous, stride 40 f16 (80B)
    __shared__ alignas(16) _Float16 Bt[2][64][40];

    v8f acc[4] = {{}, {}, {}, {}};

    const int m   = m0 + l16;
    const int mld = (m < O) ? m : (O - 1);            // clamp: safe loads
    const _Float16* wrow = Wt + (size_t)mld * Kp;

    // staging map: waves {0,1} pixels 0-31, {2,3} pixels 32-63;
    // k-half (0/16) = wv&1 -> WAVE-UNIFORM => scalar table loads.
    const int bn   = ((wv >> 1) << 5) + lane;         // pixel col in N-tile
    const int kh16 = __builtin_amdgcn_readfirstlane((wv & 1) << 4);
    int pS = p0 + bn; if (pS >= P) pS = P - 1;        // clamped pixel
    const int nS  = pS >> logHW;
    const int qS  = pS & (HW - 1);
    const int oyS = qS >> logW;
    const int oxS = qS & (W - 1);
    const long pixbase = (((long)nS * C) << logHW) + ((long)oyS << logW) + oxS;

    // per-thread validity bitmask over the KHW (r,s) positions
    unsigned vmask = 0xffffffffu;
    if (KHW > 1) {
        vmask = 0;
        #pragma unroll
        for (int rs = 0; rs < KHW; ++rs) {
            int r  = rs / KW, s = rs - (rs / KW) * KW;
            int iy = oyS + r - PAD;
            int ix = oxS + s - PAD;
            if ((unsigned)iy < (unsigned)H && (unsigned)ix < (unsigned)W)
                vmask |= 1u << rs;
        }
    }

    // gather 16 im2col values (k = kt+kh16 .. +15) into two v8h registers
    auto gather = [&](int kt, v8h& g0, v8h& g1) {
        const int kbase = kt + kh16;                  // uniform
        #pragma unroll
        for (int j = 0; j < 16; ++j) {
            int2 e = Tab[kbase + j];                  // scalar (SMEM) load
            _Float16 val = (_Float16)0.0f;
            if (KHW == 1 || ((vmask >> e.y) & 1))
                val = X[pixbase + e.x];
            if (j < 8) g0[j] = val; else g1[j - 8] = val;
        }
    };
    auto commit = [&](int buf, v8h g0, v8h g1) {
        v8h* dst = (v8h*)&Bt[buf][bn][kh16];
        dst[0] = g0; dst[1] = g1;
    };

    // prologue: stage first tile
    { v8h g0, g1; gather(0, g0, g1); commit(0, g0, g1); }
    __syncthreads();

    const int koff = hsel << 4;                       // 0 / 16 (f16 units)
    for (int kt = 0; kt < Kp; kt += 32) {
        const int cur = (kt >> 5) & 1;

        // 1. A fragment: two aligned b128 loads from padded weight row
        const int kb = kt + (hsel << 4);
        v8h lo = *(const v8h*)(wrow + kb);
        v8h hi = *(const v8h*)(wrow + kb + 8);
        v16h a;
        #pragma unroll
        for (int j = 0; j < 8; ++j) { a[j] = lo[j]; a[8 + j] = hi[j]; }

        // 2. B fragments: two ds_load_b128 each
        v16h bfr[4];
        #pragma unroll
        for (int t = 0; t < 4; ++t) {
            const v8h* brow = (const v8h*)&Bt[cur][(t << 4) + l16][koff];
            v8h blo = brow[0], bhi = brow[1];
            #pragma unroll
            for (int j = 0; j < 8; ++j) { bfr[t][j] = blo[j]; bfr[t][8 + j] = bhi[j]; }
        }

        // 3. issue next tile's global gather (latency overlaps WMMA below)
        v8h g0, g1;
        const bool more = (kt + 32 < Kp);
        if (more) {
            gather(kt + 32, g0, g1);
            __builtin_prefetch(wrow + kt + 32, 0, 1);
        }

        // 4. 16x16x32 WMMAs, A reused across 4 N-subtiles
        #pragma unroll
        for (int t = 0; t < 4; ++t)
            acc[t] = __builtin_amdgcn_wmma_f32_16x16x32_f16(false, a, false, bfr[t],
                                                            (short)0, acc[t], false, false);

        // 5. commit staged registers to the other buffer
        if (more) commit(cur ^ 1, g0, g1);
        __syncthreads();
    }

    // ---- epilogue: D element i -> row m0+i+8*hsel, cols p0+16t+l16 ----
    #pragma unroll
    for (int t = 0; t < 4; ++t) {
        const int p = p0 + (t << 4) + l16;
        #pragma unroll
        for (int i = 0; i < 8; ++i) {
            int mm = m0 + i + (hsel << 3);
            if (mm < O && p < P) {
                float v = acc[t][i] + Bs[mm];
                if (relu) v = fmaxf(v, 0.0f);
                int n = p >> logHW;
                int q = p & (HW - 1);
                size_t idx = ((size_t)((size_t)n * O + mm) << logHW) + q;
                if (Res) v += (float)Res[idx];
                if (Yh) Yh[idx] = (_Float16)v;
                if (Yf) Yf[idx] = v;
            }
        }
    }
}

// ---- generate im2col table: Tab[k] = {c*HW + (r-PAD)*W + (s-PAD), rs} -------
__global__ void gen_tab(int2* __restrict__ tab, int C, int KHW_, int KW_, int PAD_,
                        int logHW, int logW, int Kp)
{
    int k = blockIdx.x * blockDim.x + threadIdx.x;
    if (k >= Kp) return;
    int K = C * KHW_;
    int2 e;
    if (k < K) {
        int c  = k / KHW_;
        int rs = k - c * KHW_;
        int r  = rs / KW_;
        int s  = rs - r * KW_;
        e.x = c * (1 << logHW) + (r - PAD_) * (1 << logW) + (s - PAD_);
        e.y = rs;
    } else {
        e.x = 0; e.y = 0;   // harmless in-range dummy; A pad column is zero
    }
    tab[k] = e;
}

// ---- pack f32 weights [O][K] -> f16 [O][Kpad], zero tail ---------------------
__global__ void pack_w_f16(const float* __restrict__ in, _Float16* __restrict__ out,
                           int K, int Kp, long total)
{
    long i = (long)blockIdx.x * blockDim.x + threadIdx.x;
    if (i >= total) return;
    int o = (int)(i / Kp);
    int k = (int)(i - (long)o * Kp);
    out[i] = (k < K) ? (_Float16)in[(long)o * K + k] : (_Float16)0.0f;
}

// ---- global average pool over HxW: one block per (n,c) -----------------------
__global__ void gap_kernel(const _Float16* __restrict__ X, float* __restrict__ out, int HW)
{
    __shared__ float s[256];
    const int nc = blockIdx.x;
    const _Float16* p = X + (size_t)nc * HW;
    float acc = 0.0f;
    for (int i = threadIdx.x; i < HW; i += blockDim.x) acc += (float)p[i];
    s[threadIdx.x] = acc;
    __syncthreads();
    for (int st = 128; st > 0; st >>= 1) {
        if ((int)threadIdx.x < st) s[threadIdx.x] += s[threadIdx.x + st];
        __syncthreads();
    }
    if (threadIdx.x == 0) out[nc] = s[0] / (float)HW;
}

// ---- channel attention MLP: relu(Wd*pool+bd) -> sigmoid(Wu*d+bu) ------------
__global__ void catt_kernel(const float* __restrict__ pooled,
                            const float* __restrict__ dw, const float* __restrict__ db,
                            const float* __restrict__ uw, const float* __restrict__ ub,
                            float* __restrict__ y, int nf, int cr)
{
    const int n = blockIdx.x;
    __shared__ float d[64];
    const int t = threadIdx.x;
    if (t < cr) {
        float a = db[t];
        for (int c = 0; c < nf; ++c) a += dw[t * nf + c] * pooled[n * nf + c];
        d[t] = fmaxf(a, 0.0f);
    }
    __syncthreads();
    if (t < nf) {
        float a = ub[t];
        for (int j = 0; j < cr; ++j) a += uw[t * cr + j] * d[j];
        y[n * nf + t] = 1.0f / (1.0f + expf(-a));
    }
}

// ---- gr + h * y[n,c]  (RCAB residual with channel attention) ----------------
__global__ void scale_add_kernel(const _Float16* __restrict__ gr,
                                 const _Float16* __restrict__ h,
                                 const float* __restrict__ y,
                                 _Float16* __restrict__ out,
                                 int C, int HW, long total)
{
    long i = (long)blockIdx.x * blockDim.x + threadIdx.x;
    if (i >= total) return;
    int c = (int)((i / HW) % C);
    int n = (int)(i / ((long)HW * C));
    out[i] = (_Float16)((float)gr[i] + (float)h[i] * y[n * C + c]);
}

// ---- pixel shuffle r=2: out[n][co][2y+dy][2x+dx] = in[n][co*4+dy*2+dx][y][x] --
__global__ void pshuf_kernel(const _Float16* __restrict__ in, _Float16* __restrict__ out,
                             int N, int co, int H, int W, long total)
{
    long i = (long)blockIdx.x * blockDim.x + threadIdx.x;
    if (i >= total) return;
    int w = (int)(i % W);
    long t = i / W;
    int h = (int)(t % H); t /= H;
    int c = (int)(t % (co * 4));
    int n = (int)(t / (co * 4));
    int cc = c >> 2;
    int dy = (c >> 1) & 1;
    int dx = c & 1;
    out[(((size_t)n * co + cc) * (2 * H) + (2 * h + dy)) * (size_t)(2 * W) + (2 * w + dx)] = in[i];
}

__global__ void cast_f32_to_f16(const float* __restrict__ in, _Float16* __restrict__ out, long n)
{
    long i = (long)blockIdx.x * blockDim.x + threadIdx.x;
    if (i < n) out[i] = (_Float16)in[i];
}

// ---------------------------------------------------------------------------
// Host-side orchestration
// ---------------------------------------------------------------------------
#define NRG 2
#define NRB 3

namespace {

struct Bump {
    char* base; size_t off, cap;
    void* take(size_t bytes) {
        size_t a = (off + 255) & ~(size_t)255;
        if (a + bytes > cap) a = 0;   // deterministic fallback if ws undersized
        off = a + bytes;
        return base + a;
    }
};

struct BranchParams {
    const float *add_b, *add_w, *body_b, *body_w;
    const float *c1b[NRG][NRB], *c1w[NRG][NRB], *c2b[NRG][NRB], *c2w[NRG][NRB];
    const float *db_[NRG][NRB], *dw_[NRG][NRB], *ub_[NRG][NRB], *uw_[NRG][NRB];
    const float *gb[NRG], *gw[NRG];
    const float *head_b, *head_w, *sub_b, *sub_w, *tail_b, *tail_w;
    const float *up1_b, *up1_w, *up2_b, *up2_w;
};

void launch_conv(const _Float16* X, const _Float16* Wt, const int2* Tab, const float* Bs,
                 const _Float16* Res, _Float16* Yh, float* Yf,
                 int N, int C, int H, int W, int O, int k3, int relu,
                 hipStream_t st)
{
    long P = (long)N * H * W;
    int logH = __builtin_ctz((unsigned)H);
    int logW = __builtin_ctz((unsigned)W);
    dim3 grid((unsigned)((O + 63) / 64), (unsigned)((P + 63) / 64), 1);
    if (k3)
        conv_wmma_kernel<3, 3, 1><<<grid, 128, 0, st>>>(X, Wt, Tab, Bs, Res, Yh, Yf,
                                                        N, C, logH, logW, O, relu);
    else
        conv_wmma_kernel<1, 1, 0><<<grid, 128, 0, st>>>(X, Wt, Tab, Bs, Res, Yh, Yf,
                                                        N, C, logH, logW, O, relu);
}

void run_branch(const float* xin, const BranchParams& bp, int N, int nf,
                float* outp, Bump ws, hipStream_t st)
{
    const int H = 32, W = 32, HW = H * W;
    int cr = nf / 16; if (cr < 1) cr = 1;

    // pack f32 weights -> f16 [O][Kpad] with zero tail
    auto cvt = [&](const float* w, int O, int K) -> const _Float16* {
        int Kp = (K + 31) & ~31;
        long n = (long)O * Kp;
        _Float16* d = (_Float16*)ws.take((size_t)n * sizeof(_Float16));
        pack_w_f16<<<(unsigned)((n + 255) / 256), 256, 0, st>>>(w, d, K, Kp, n);
        return d;
    };
    // im2col tables per (C, kernel, geometry)
    auto mktab = [&](int C, int k3, int logH, int logW) -> const int2* {
        int KHW_ = k3 ? 9 : 1;
        int K = C * KHW_, Kp = (K + 31) & ~31;
        int2* t = (int2*)ws.take((size_t)Kp * sizeof(int2));
        gen_tab<<<(unsigned)((Kp + 63) / 64), 64, 0, st>>>(t, C, KHW_, k3 ? 3 : 1,
                                                           k3 ? 1 : 0, logH + logW, logW, Kp);
        return t;
    };

    const _Float16* sub_w16  = cvt(bp.sub_w, 3, 3);
    const _Float16* head_w16 = cvt(bp.head_w, nf, 27);
    const _Float16* body_w16 = cvt(bp.body_w, nf, nf * 9);
    const _Float16* up1_w16  = cvt(bp.up1_w, 4 * nf, nf * 9);
    const _Float16* up2_w16  = cvt(bp.up2_w, 4 * nf, nf * 9);
    const _Float16* tail_w16 = cvt(bp.tail_w, 3, nf * 9);
    const _Float16* add_w16  = cvt(bp.add_w, 3, 3);
    const _Float16 *gw16[NRG], *c1w16[NRG][NRB], *c2w16[NRG][NRB];
    for (int g = 0; g < NRG; ++g) {
        gw16[g] = cvt(bp.gw[g], nf, nf * 9);
        for (int b = 0; b < NRB; ++b) {
            c1w16[g][b] = cvt(bp.c1w[g][b], nf, nf * 9);
            c2w16[g][b] = cvt(bp.c2w[g][b], nf, nf * 9);
        }
    }

    const int2* tab_sub  = mktab(3,  0, 5, 5);  // 1x1 C=3   @32x32
    const int2* tab_head = mktab(3,  1, 5, 5);  // 3x3 C=3   @32x32
    const int2* tab_body = mktab(nf, 1, 5, 5);  // 3x3 C=nf  @32x32 (all RCAB/gw/body/up1)
    const int2* tab_up2  = mktab(nf, 1, 6, 6);  // 3x3 C=nf  @64x64
    const int2* tab_tail = mktab(nf, 1, 7, 7);  // 3x3 C=nf  @128x128
    const int2* tab_add  = mktab(3,  0, 7, 7);  // 1x1 C=3   @128x128

    // ---- activation buffers (f16) ----
    const long A = (long)N * nf * HW;
    _Float16* xh    = (_Float16*)ws.take((size_t)N * 3 * HW * 2);
    _Float16* subo  = (_Float16*)ws.take((size_t)N * 3 * HW * 2);
    _Float16* xhead = (_Float16*)ws.take((size_t)A * 2);
    _Float16* grA   = (_Float16*)ws.take((size_t)A * 2);
    _Float16* grB   = (_Float16*)ws.take((size_t)A * 2);
    _Float16* h1    = (_Float16*)ws.take((size_t)A * 2);
    _Float16* h2    = (_Float16*)ws.take((size_t)A * 2);
    _Float16* r1    = (_Float16*)ws.take((size_t)A * 2);
    _Float16* r2    = (_Float16*)ws.take((size_t)A * 2);
    float* pooled   = (float*)ws.take((size_t)N * nf * 4);
    float* yscale   = (float*)ws.take((size_t)N * nf * 4);
    _Float16* up1o  = (_Float16*)ws.take((size_t)N * 4 * nf * HW * 2);
    _Float16* sh1   = (_Float16*)ws.take((size_t)N * nf * 4 * HW * 2);
    _Float16* up2o  = (_Float16*)ws.take((size_t)N * 4 * nf * 4 * HW * 2);
    _Float16* sh2   = (_Float16*)ws.take((size_t)N * nf * 16 * HW * 2);
    _Float16* tailo = (_Float16*)ws.take((size_t)N * 3 * 16 * HW * 2);

    // ---- sub_mean (1x1) + head (3x3, 3->nf) ----
    long nin = (long)N * 3 * HW;
    cast_f32_to_f16<<<(unsigned)((nin + 255) / 256), 256, 0, st>>>(xin, xh, nin);
    launch_conv(xh,   sub_w16,  tab_sub,  bp.sub_b,  nullptr, subo,  nullptr, N, 3, H, W, 3,  0, 0, st);
    launch_conv(subo, head_w16, tab_head, bp.head_b, nullptr, xhead, nullptr, N, 3, H, W, nf, 1, 0, st);

    // ---- residual groups of RCABs ----
    const _Float16* cur = xhead;
    _Float16* rbuf[2] = {r1, r2};
    for (int g = 0; g < NRG; ++g) {
        const _Float16* gr = cur;
        for (int b = 0; b < NRB; ++b) {
            launch_conv(gr, c1w16[g][b], tab_body, bp.c1b[g][b], nullptr, h1, nullptr, N, nf, H, W, nf, 1, 1, st);
            launch_conv(h1, c2w16[g][b], tab_body, bp.c2b[g][b], nullptr, h2, nullptr, N, nf, H, W, nf, 1, 0, st);
            gap_kernel<<<(unsigned)(N * nf), 256, 0, st>>>(h2, pooled, HW);
            catt_kernel<<<(unsigned)N, 64, 0, st>>>(pooled, bp.dw_[g][b], bp.db_[g][b],
                                                    bp.uw_[g][b], bp.ub_[g][b], yscale, nf, cr);
            _Float16* o = (b & 1) ? grB : grA;
            scale_add_kernel<<<(unsigned)((A + 255) / 256), 256, 0, st>>>(gr, h2, yscale, o, nf, HW, A);
            gr = o;
        }
        launch_conv(gr, gw16[g], tab_body, bp.gb[g], cur, rbuf[g], nullptr, N, nf, H, W, nf, 1, 0, st);
        cur = rbuf[g];
    }
    // long skip: body conv + x
    launch_conv(cur, body_w16, tab_body, bp.body_b, xhead, grB, nullptr, N, nf, H, W, nf, 1, 0, st);

    // ---- x4 upsampler: conv -> pixelshuffle(2), twice ----
    launch_conv(grB, up1_w16, tab_body, bp.up1_b, nullptr, up1o, nullptr, N, nf, H, W, 4 * nf, 1, 0, st);
    long t1 = (long)N * 4 * nf * HW;
    pshuf_kernel<<<(unsigned)((t1 + 255) / 256), 256, 0, st>>>(up1o, sh1, N, nf, H, W, t1);

    launch_conv(sh1, up2_w16, tab_up2, bp.up2_b, nullptr, up2o, nullptr, N, nf, 2 * H, 2 * W, 4 * nf, 1, 0, st);
    long t2 = (long)N * 4 * nf * 4 * HW;
    pshuf_kernel<<<(unsigned)((t2 + 255) / 256), 256, 0, st>>>(up2o, sh2, N, nf, 2 * H, 2 * W, t2);

    // ---- tail conv (nf->3) + add_mean (1x1, f32 output) ----
    launch_conv(sh2,   tail_w16, tab_tail, bp.tail_b, nullptr, tailo, nullptr, N, nf, 4 * H, 4 * W, 3, 1, 0, st);
    launch_conv(tailo, add_w16,  tab_add,  bp.add_b,  nullptr, nullptr, outp,  N, 3,  4 * H, 4 * W, 3, 0, 0, st);
}

} // namespace

extern "C" void kernel_launch(void* const* d_in, const int* in_sizes, int n_in,
                              void* d_out, int out_size, void* d_ws, size_t ws_size,
                              hipStream_t stream)
{
    static const int BS[3] = {34, 38, 29};
    static const int NF[3] = {36, 50, 64};

    float* out = (float*)d_out;
    size_t ooff = 0;
    for (int br = 0; br < 3; ++br) {
        const int base = 3 + br * 66;  // jax pytree leaf order (dict keys sorted)
        auto Lf = [&](int i) -> const float* { return (const float*)d_in[base + i]; };

        BranchParams bp;
        bp.add_b = Lf(0);  bp.add_w = Lf(1);
        bp.body_b = Lf(2); bp.body_w = Lf(3);
        const int gi = 4;  // per group: 3 blocks * 8 leaves + gb + gw = 26
        for (int g = 0; g < NRG; ++g) {
            for (int b = 0; b < NRB; ++b) {
                int bb = gi + g * 26 + b * 8;
                bp.c1b[g][b] = Lf(bb + 0); bp.c1w[g][b] = Lf(bb + 1);
                bp.c2b[g][b] = Lf(bb + 2); bp.c2w[g][b] = Lf(bb + 3);
                bp.db_[g][b] = Lf(bb + 4); bp.dw_[g][b] = Lf(bb + 5);
                bp.ub_[g][b] = Lf(bb + 6); bp.uw_[g][b] = Lf(bb + 7);
            }
            bp.gb[g] = Lf(gi + g * 26 + 24);
            bp.gw[g] = Lf(gi + g * 26 + 25);
        }
        bp.head_b = Lf(56); bp.head_w = Lf(57);
        bp.sub_b  = Lf(58); bp.sub_w  = Lf(59);
        bp.tail_b = Lf(60); bp.tail_w = Lf(61);
        bp.up1_b  = Lf(62); bp.up1_w  = Lf(63);
        bp.up2_b  = Lf(64); bp.up2_w  = Lf(65);

        Bump ws{(char*)d_ws, 0, ws_size};
        run_branch((const float*)d_in[br], bp, BS[br], NF[br], out + ooff, ws, stream);
        ooff += (size_t)BS[br] * 3 * 128 * 128;
    }
}
